// TransformerEncoder_9998683865337
// MI455X (gfx1250) — compile-verified, matrix operand order checked by reading, not run on hardware
//
#include <hip/hip_runtime.h>
#include <hip/hip_bf16.h>
#include <math.h>

// ---------------------------------------------------------------------------
// Transformer encoder layer for MI455X (gfx1250), wave32 + WMMA f16->f32,
// with async-to-LDS double-buffered GEMM staging (ASYNCcnt pipeline).
// B=8, S=1024, D=1024, H=16, DH=64, DFF=4096.
// ---------------------------------------------------------------------------

typedef _Float16 half_t;
typedef __attribute__((ext_vector_type(8)))  _Float16 v8h;
typedef __attribute__((ext_vector_type(16))) _Float16 v16h;
typedef __attribute__((ext_vector_type(8)))  float    v8f;

#define CB  8
#define CS  1024
#define CD  1024
#define CH  16
#define CDH 64
#define CDFF 4096
#define CBS (CB * CS)
#define INV_SCALE 0.125f   // 1/sqrt(64)

#define LDS_STRIDE 40      // halves per LDS tile row (32 data + 8 pad) -> conflict-free b128
#define TILE_HALFS (128 * LDS_STRIDE)

// Load one A/B WMMA fragment: halves 0..7 from p0, 8..15 from p1 (two b128s).
__device__ __forceinline__ v16h ld_frag(const half_t* p0, const half_t* p1) {
    v8h lo = *(const v8h*)p0;
    v8h hi = *(const v8h*)p1;
    return __builtin_shufflevector(lo, hi, 0,1,2,3,4,5,6,7,8,9,10,11,12,13,14,15);
}

__device__ __forceinline__ v8f wmma_f16(v16h a, v16h b, v8f c) {
    return __builtin_amdgcn_wmma_f32_16x16x32_f16(false, a, false, b, (short)0, c,
                                                  false, false);
}

// Async global->LDS copy of 16 bytes per lane (ASYNCcnt-tracked, no VGPRs).
__device__ __forceinline__ void async_copy_b128(const half_t* lds_ptr,
                                                const half_t* gptr) {
    unsigned lds_off = (unsigned)(size_t)lds_ptr;  // addr[31:0] = LDS byte offset
    asm volatile("global_load_async_to_lds_b128 %0, %1, off"
                 :: "v"(lds_off), "v"(gptr) : "memory");
}
__device__ __forceinline__ void wait_async_le4() {
    asm volatile("s_wait_asynccnt 0x4" ::: "memory");
}
__device__ __forceinline__ void wait_async_0() {
    asm volatile("s_wait_asynccnt 0x0" ::: "memory");
}

// Stage A[128 x 32] and Bt[128 x 32] K-chunk tiles into LDS (4 async b128/thread).
// Buffers addressed as integer offsets into the kernel's __shared__ array so
// the compute-side reads keep addrspace(3) and lower to ds_load_b128.
__device__ __forceinline__ void stage_tiles(const half_t* __restrict__ A,
                                            const half_t* __restrict__ Bt,
                                            int K, int rowb, int colb, int kc,
                                            half_t* smem, int asBase, int bsBase,
                                            int t) {
#pragma unroll
    for (int p = 0; p < 2; ++p) {
        int u = t + 256 * p;            // 512 units of 16B per 8KB tile
        int row = u >> 2, cu = u & 3;   // 4 x 16B per 64B tile row
        async_copy_b128(&smem[asBase + row * LDS_STRIDE + cu * 8],
                        A + (size_t)(rowb + row) * K + kc + cu * 8);
        async_copy_b128(&smem[bsBase + row * LDS_STRIDE + cu * 8],
                        Bt + (size_t)(colb + row) * K + kc + cu * 8);
    }
}

// Wave computes its 64(M) x 32(N) sub-tile for one K=32 chunk from LDS.
__device__ __forceinline__ void compute_chunk(const half_t* smem, int asBase, int bsBase,
                                              int waveM, int waveN, int m, int hi,
                                              v8f acc[4][2]) {
    v16h bf[2];
#pragma unroll
    for (int j = 0; j < 2; ++j) {
        const half_t* p = &smem[bsBase + (waveN * 32 + 16 * j + m) * LDS_STRIDE + 8 * hi];
        bf[j] = ld_frag(p, p + 16);
    }
#pragma unroll
    for (int i = 0; i < 4; ++i) {
        const half_t* p = &smem[asBase + (waveM * 64 + 16 * i + m) * LDS_STRIDE + 8 * hi];
        v16h af = ld_frag(p, p + 16);
#pragma unroll
        for (int j = 0; j < 2; ++j)
            acc[i][j] = wmma_f16(af, bf[j], acc[i][j]);
    }
}

// Full double-buffered 128x128-block GEMM accumulation (call from kernel body).
__device__ __forceinline__ void gemm_block_lds(const half_t* __restrict__ A,
                                               const half_t* __restrict__ Bt,
                                               int K, int rowb, int colb,
                                               half_t* smem, int t,
                                               int waveM, int waveN, int m, int hi,
                                               v8f acc[4][2]) {
    const int nk = K / 32;
    stage_tiles(A, Bt, K, rowb, colb, 0, smem, 0, TILE_HALFS, t);
    for (int c = 0; c < nk; ++c) {
        const int cur = c & 1;
        const int asCur = cur ? 2 * TILE_HALFS : 0;
        const int bsCur = asCur + TILE_HALFS;
        if (c + 1 < nk) {
            const int asNxt = cur ? 0 : 2 * TILE_HALFS;
            stage_tiles(A, Bt, K, rowb, colb, (c + 1) * 32, smem,
                        asNxt, asNxt + TILE_HALFS, t);
            wait_async_le4();   // retire chunk c's copies; chunk c+1 still in flight
        } else {
            wait_async_0();
        }
        __syncthreads();
        compute_chunk(smem, asCur, bsCur, waveM, waveN, m, hi, acc);
        __syncthreads();        // everyone done reading before buffer is re-filled
    }
}

// ------------------------------ LayerNorm ----------------------------------
__global__ __launch_bounds__(256)
void layernorm_f16_kernel(const float* __restrict__ x, const float* __restrict__ g,
                          const float* __restrict__ be, half_t* __restrict__ out) {
    __shared__ float rs[256], rs2[256];
    const int row = blockIdx.x, t = threadIdx.x;
    const float* xr = x + (size_t)row * CD;
    float s = 0.f, s2 = 0.f;
#pragma unroll
    for (int i = t; i < CD; i += 256) { float v = xr[i]; s += v; s2 += v * v; }
    rs[t] = s; rs2[t] = s2; __syncthreads();
    for (int o = 128; o > 0; o >>= 1) {
        if (t < o) { rs[t] += rs[t + o]; rs2[t] += rs2[t + o]; }
        __syncthreads();
    }
    const float mu   = rs[0] * (1.f / CD);
    const float var  = rs2[0] * (1.f / CD) - mu * mu;
    const float rstd = rsqrtf(var + 1e-5f);
#pragma unroll
    for (int i = t; i < CD; i += 256)
        out[(size_t)row * CD + i] = (half_t)((xr[i] - mu) * rstd * g[i] + be[i]);
}

// --------------------------- weight conversion -----------------------------
// Wt[n][k] = W[k][n], fp32 -> f16.  W is [K x N] row-major.
__global__ __launch_bounds__(256)
void transpose_to_f16_kernel(const float* __restrict__ W, half_t* __restrict__ Wt,
                             int K, int N) {
    size_t idx = (size_t)blockIdx.x * 256 + threadIdx.x;
    if (idx >= (size_t)K * N) return;
    int n = (int)(idx / K), k = (int)(idx % K);
    Wt[idx] = (half_t)W[(size_t)k * N + n];
}

// Wqkv_t [3*D][D]: row n -> (which = n/1024, h = (n%1024)/64, kk = n%64),
// value = W{q,k,v}[h][d][kk] laid out [H, D, DH].
__global__ __launch_bounds__(256)
void build_wqkv_t_kernel(const float* __restrict__ Wq, const float* __restrict__ Wk,
                         const float* __restrict__ Wv, half_t* __restrict__ Wt) {
    size_t idx = (size_t)blockIdx.x * 256 + threadIdx.x;   // 3*D*D
    int n = (int)(idx >> 10), d = (int)(idx & 1023);
    int which = n >> 10, cc = n & 1023, h = cc >> 6, kk = cc & 63;
    const float* W = (which == 0) ? Wq : (which == 1) ? Wk : Wv;
    Wt[idx] = (half_t)W[((size_t)h * CD + d) * CDH + kk];
}

// ------------------------------- QKV GEMM ----------------------------------
// n1[BS x D] * Wqkv_t^T -> scatter to q/k [B,H,S,DH] f16 and v^T [B,H,DH,S] f16.
__global__ __launch_bounds__(256)
void gemm_qkv_kernel(const half_t* __restrict__ A, const half_t* __restrict__ Bt,
                     const float* __restrict__ bq, const float* __restrict__ bk,
                     const float* __restrict__ bv, half_t* __restrict__ qh,
                     half_t* __restrict__ kh, half_t* __restrict__ vt) {
    __shared__ __align__(16) half_t smem[4 * TILE_HALFS];
    const int t = threadIdx.x, lane = t & 31, wave = t >> 5;
    const int waveM = wave & 1, waveN = wave >> 1;
    const int row0 = blockIdx.y * 128 + waveM * 64;
    const int col0 = blockIdx.x * 128 + waveN * 32;
    const int m = lane & 15, hi = lane >> 4;
    v8f acc[4][2] = {};
    gemm_block_lds(A, Bt, CD, blockIdx.y * 128, blockIdx.x * 128, smem, t,
                   waveM, waveN, m, hi, acc);
#pragma unroll
    for (int i = 0; i < 4; ++i)
#pragma unroll
        for (int j = 0; j < 2; ++j)
#pragma unroll
            for (int r = 0; r < 8; ++r) {
                int row = row0 + 16 * i + r + 8 * hi;      // b*S + s
                int col = col0 + 16 * j + m;               // 0..3071
                int b = row >> 10, s = row & 1023;
                int which = col >> 10, cc = col & 1023, h = cc >> 6, dh = cc & 63;
                float bias = (which == 0) ? bq[cc] : (which == 1) ? bk[cc] : bv[cc];
                float val = acc[i][j][r] + bias;
                if (which == 2)
                    vt[((size_t)(b * CH + h) * CDH + dh) * CS + s] = (half_t)val;
                else {
                    half_t* dst = which ? kh : qh;
                    dst[((size_t)(b * CH + h) * CS + s) * CDH + dh] = (half_t)val;
                }
            }
}

// --------------------------- flash attention -------------------------------
// One wave = 16 queries. Scores computed transposed (M=kv, N=q) so the softmax
// row reduction is per-lane + one shfl_xor(16); P^T repacks in registers into
// the next WMMA's B fragment. O^T = V^T * P^T with online rescale.
__global__ __launch_bounds__(128)
void attn_kernel(const half_t* __restrict__ qh, const half_t* __restrict__ kh,
                 const half_t* __restrict__ vt, half_t* __restrict__ oh) {
    const int lane = threadIdx.x & 31, wave = threadIdx.x >> 5;
    const int bh = blockIdx.x;                       // b*H + h
    const int b = bh >> 4, h = bh & 15;
    const int q0 = (blockIdx.y * 4 + wave) * 16;
    const int m = lane & 15, hi = lane >> 4;

    const half_t* qbase = qh + (size_t)bh * CS * CDH;
    const half_t* kbase = kh + (size_t)bh * CS * CDH;
    const half_t* vbase = vt + (size_t)bh * CDH * CS;

    v16h qf[2];
#pragma unroll
    for (int c = 0; c < 2; ++c) {
        const half_t* p = qbase + (size_t)(q0 + m) * CDH + 32 * c + 8 * hi;
        qf[c] = ld_frag(p, p + 16);
    }

    float m_i = -1e30f, l_i = 0.f;
    v8f o[4] = {};

    for (int kv = 0; kv < CS; kv += 32) {
        v8f s0 = {}, s1 = {};
#pragma unroll
        for (int c = 0; c < 2; ++c) {
            const half_t* p0 = kbase + (size_t)(kv + m) * CDH + 32 * c + 8 * hi;
            const half_t* p1 = kbase + (size_t)(kv + 16 + m) * CDH + 32 * c + 8 * hi;
            s0 = wmma_f16(ld_frag(p0, p0 + 16), qf[c], s0);
            s1 = wmma_f16(ld_frag(p1, p1 + 16), qf[c], s1);
        }
        float mx = -1e30f;
#pragma unroll
        for (int r = 0; r < 8; ++r) {
            s0[r] *= INV_SCALE; s1[r] *= INV_SCALE;
            mx = fmaxf(mx, fmaxf(s0[r], s1[r]));
        }
        mx = fmaxf(mx, __shfl_xor(mx, 16, 32));
        const float m_new = fmaxf(m_i, mx);
        const float alpha = __expf(m_i - m_new);
        float p0v[8], p1v[8], psum = 0.f;
#pragma unroll
        for (int r = 0; r < 8; ++r) {
            p0v[r] = __expf(s0[r] - m_new);
            p1v[r] = __expf(s1[r] - m_new);
            psum += p0v[r] + p1v[r];
        }
        psum += __shfl_xor(psum, 16, 32);
        l_i = l_i * alpha + psum;
        m_i = m_new;
#pragma unroll
        for (int tt = 0; tt < 4; ++tt)
#pragma unroll
            for (int r = 0; r < 8; ++r) o[tt][r] *= alpha;
        v16h pf;
#pragma unroll
        for (int r = 0; r < 8; ++r) {
            pf[r]     = (half_t)p0v[r];
            pf[8 + r] = (half_t)p1v[r];
        }
#pragma unroll
        for (int tt = 0; tt < 4; ++tt) {
            const half_t* p = vbase + (size_t)(16 * tt + m) * CS + kv + 8 * hi;
            o[tt] = wmma_f16(ld_frag(p, p + 16), pf, o[tt]);
        }
    }
    const float inv_l = 1.f / l_i;
#pragma unroll
    for (int tt = 0; tt < 4; ++tt)
#pragma unroll
        for (int r = 0; r < 8; ++r) {
            int dh = 16 * tt + r + 8 * hi;
            size_t idx = ((size_t)(b * CS + q0 + m)) * CD + h * CDH + dh;
            oh[idx] = (half_t)(o[tt][r] * inv_l);
        }
}

// -------------------------- fused GEMM epilogues ---------------------------
// mode 0: outF = acc + bias + addsrc (residual), f32 out
// mode 1: outH = gelu(acc + bias), f16 out
__global__ __launch_bounds__(256)
void gemm_bias_act_kernel(const half_t* __restrict__ A, const half_t* __restrict__ Bt,
                          const float* __restrict__ bias, const float* __restrict__ addsrc,
                          float* __restrict__ outF, half_t* __restrict__ outH,
                          int N, int K, int mode) {
    __shared__ __align__(16) half_t smem[4 * TILE_HALFS];
    const int t = threadIdx.x, lane = t & 31, wave = t >> 5;
    const int waveM = wave & 1, waveN = wave >> 1;
    const int row0 = blockIdx.y * 128 + waveM * 64;
    const int col0 = blockIdx.x * 128 + waveN * 32;
    const int m = lane & 15, hi = lane >> 4;
    v8f acc[4][2] = {};
    gemm_block_lds(A, Bt, K, blockIdx.y * 128, blockIdx.x * 128, smem, t,
                   waveM, waveN, m, hi, acc);
#pragma unroll
    for (int i = 0; i < 4; ++i)
#pragma unroll
        for (int j = 0; j < 2; ++j)
#pragma unroll
            for (int r = 0; r < 8; ++r) {
                int row = row0 + 16 * i + r + 8 * hi;
                int col = col0 + 16 * j + m;
                size_t idx = (size_t)row * N + col;
                float v = acc[i][j][r] + bias[col];
                if (mode == 0) {
                    outF[idx] = v + addsrc[idx];
                } else {
                    v = 0.5f * v * (1.f + erff(v * 0.70710678118654752f));
                    outH[idx] = (half_t)v;
                }
            }
}

// ------------------------------- launcher ----------------------------------
extern "C" void kernel_launch(void* const* d_in, const int* in_sizes, int n_in,
                              void* d_out, int out_size, void* d_ws, size_t ws_size,
                              hipStream_t stream) {
    const float* x   = (const float*)d_in[0];
    const float* Wq  = (const float*)d_in[1];
    const float* bq  = (const float*)d_in[2];
    const float* Wk  = (const float*)d_in[3];
    const float* bk  = (const float*)d_in[4];
    const float* Wv  = (const float*)d_in[5];
    const float* bv  = (const float*)d_in[6];
    const float* Wp  = (const float*)d_in[7];
    const float* bp  = (const float*)d_in[8];
    const float* g1  = (const float*)d_in[9];
    const float* be1 = (const float*)d_in[10];
    const float* g2  = (const float*)d_in[11];
    const float* be2 = (const float*)d_in[12];
    const float* W1  = (const float*)d_in[13];
    const float* b1  = (const float*)d_in[14];
    const float* W2  = (const float*)d_in[15];
    const float* b2  = (const float*)d_in[16];

    // workspace carve (all offsets 256B aligned)
    char* ws = (char*)d_ws;
    size_t off = 0;
    auto carve = [&](size_t bytes) { char* p = ws + off; off += (bytes + 255) & ~(size_t)255; return p; };
    half_t* wqkv_t = (half_t*)carve((size_t)3 * CD * CD * 2);
    half_t* wp_t   = (half_t*)carve((size_t)CD * CD * 2);
    half_t* w1_t   = (half_t*)carve((size_t)CDFF * CD * 2);
    half_t* w2_t   = (half_t*)carve((size_t)CD * CDFF * 2);
    half_t* n1_h   = (half_t*)carve((size_t)CBS * CD * 2);
    half_t* q_h    = (half_t*)carve((size_t)CBS * CD * 2);
    half_t* k_h    = (half_t*)carve((size_t)CBS * CD * 2);
    half_t* vt_h   = (half_t*)carve((size_t)CBS * CD * 2);
    half_t* o_h    = (half_t*)carve((size_t)CBS * CD * 2);
    float*  r1     = (float*) carve((size_t)CBS * CD * 4);
    half_t* n2_h   = (half_t*)carve((size_t)CBS * CD * 2);
    half_t* h1_h   = (half_t*)carve((size_t)CBS * CDFF * 2);
    (void)ws_size; (void)in_sizes; (void)n_in; (void)out_size;

    // 1. LN1 + weight conversions
    layernorm_f16_kernel<<<CBS, 256, 0, stream>>>(x, g1, be1, n1_h);
    build_wqkv_t_kernel<<<(3 * CD * CD) / 256, 256, 0, stream>>>(Wq, Wk, Wv, wqkv_t);
    transpose_to_f16_kernel<<<(CD * CD) / 256, 256, 0, stream>>>(Wp, wp_t, CD, CD);
    transpose_to_f16_kernel<<<(CD * CDFF) / 256, 256, 0, stream>>>(W1, w1_t, CD, CDFF);
    transpose_to_f16_kernel<<<(CDFF * CD) / 256, 256, 0, stream>>>(W2, w2_t, CDFF, CD);

    // 2. QKV projection (M=8192, N=3072, K=1024)
    gemm_qkv_kernel<<<dim3(3 * CD / 128, CBS / 128), 256, 0, stream>>>(
        n1_h, wqkv_t, bq, bk, bv, q_h, k_h, vt_h);

    // 3. attention (flash, per-wave 16 queries)
    attn_kernel<<<dim3(CB * CH, CS / 64), 128, 0, stream>>>(q_h, k_h, vt_h, o_h);

    // 4. output projection + residual -> r1
    gemm_bias_act_kernel<<<dim3(CD / 128, CBS / 128), 256, 0, stream>>>(
        o_h, wp_t, bp, x, r1, nullptr, CD, CD, 0);

    // 5. LN2
    layernorm_f16_kernel<<<CBS, 256, 0, stream>>>(r1, g2, be2, n2_h);

    // 6. MLP up + exact GELU -> f16
    gemm_bias_act_kernel<<<dim3(CDFF / 128, CBS / 128), 256, 0, stream>>>(
        n2_h, w1_t, b1, nullptr, nullptr, h1_h, CDFF, CD, 1);

    // 7. MLP down + residual -> d_out (f32)
    gemm_bias_act_kernel<<<dim3(CD / 128, CBS / 128), 256, 0, stream>>>(
        h1_h, w2_t, b2, r1, (float*)d_out, nullptr, CD, CDFF, 0);
}
